// Attention_60395830116504
// MI455X (gfx1250) — compile-verified, matrix operand order checked by reading, not run on hardware
//
#include <hip/hip_runtime.h>

// ---- shapes (compile-time, from reference) ----
constexpr int Bc = 2, Sc = 2048, Ec = 1024, Hc = 16, Dc = 64, E3 = 3 * Ec;

typedef __attribute__((ext_vector_type(16))) __bf16 v16bf;
typedef __attribute__((ext_vector_type(8)))  __bf16 v8bf;
typedef __attribute__((ext_vector_type(4)))  __bf16 v4bf;
typedef __attribute__((ext_vector_type(8)))  float  v8f;
typedef __attribute__((ext_vector_type(4)))  float  v4f;

__device__ inline v8f zero8() {
    v8f z;
#pragma unroll
    for (int i = 0; i < 8; ++i) z[i] = 0.0f;
    return z;
}

// Assemble a 16x32 bf16 A/B fragment for one lane from a row-major (K-contiguous)
// row pointer. Per ISA 7.12.2: lanes 0-15 hold K {0..7,16..23}, lanes 16-31 hold
// K {8..15,24..31}; o0/o1 are those element offsets.
__device__ inline v16bf frag16(const __bf16* p, int o0, int o1) {
    v8bf lo = *(const v8bf*)(p + o0);
    v8bf hi = *(const v8bf*)(p + o1);
    v16bf r;
#pragma unroll
    for (int i = 0; i < 8; ++i) { r[i] = lo[i]; r[i + 8] = hi[i]; }
    return r;
}

// ============================================================================
// Kernel 1: qkv = X @ W_in^T + b_in  ->  Q,K,V bf16 in [B,H,S,D]; Q pre-scaled.
// M=4096, N=3072, K=1024. Block tile 128x128, 8 waves (2M x 4N), wave = 64x32.
// Software pipelined: prefetch next k-tile into registers during WMMA burst.
// ============================================================================
__global__ __launch_bounds__(256) void qkv_gemm_kernel(
    const float* __restrict__ X, const float* __restrict__ W,
    const float* __restrict__ bias,
    __bf16* __restrict__ Q, __bf16* __restrict__ K, __bf16* __restrict__ V) {
    constexpr int BM = 128, BN = 128, BK = 32, LDT = 48;  // 96B rows, 16B chunks
    __shared__ __align__(16) __bf16 As[BM][LDT];
    __shared__ __align__(16) __bf16 Bs[BN][LDT];

    const int tid = threadIdx.x;
    const int lane = tid & 31, wv = tid >> 5;
    const int wm = wv >> 2, wn = wv & 3;   // 2 x 4 wave grid: 64 rows x 32 cols
    const int m0 = blockIdx.y * BM;
    const int f0 = blockIdx.x * BN;
    const int lrow = lane & 15, lhi = lane >> 4;
    const int o0 = lhi ? 8 : 0, o1 = lhi ? 24 : 16;

    v8f acc[2][4];                         // [nf][mf]
#pragma unroll
    for (int nf = 0; nf < 2; ++nf)
#pragma unroll
        for (int mf = 0; mf < 4; ++mf) acc[nf][mf] = zero8();

    v4f pa[4], pb[4];                      // register prefetch (A,B: 4 float4 each)
    auto globalLoad = [&](int k0) {
#pragma unroll
        for (int it = 0; it < 4; ++it) {   // 128x32 f32 = 1024 float4 / 256 thr
            int idx = tid + it * 256;
            int r = idx >> 3, c4 = idx & 7;
            pa[it] = *(const v4f*)(X + (size_t)(m0 + r) * Ec + k0 + c4 * 4);
            pb[it] = *(const v4f*)(W + (size_t)(f0 + r) * Ec + k0 + c4 * 4);
        }
    };
    auto ldsStore = [&]() {
#pragma unroll
        for (int it = 0; it < 4; ++it) {
            int idx = tid + it * 256;
            int r = idx >> 3, c4 = idx & 7;
            v4bf a, b;
#pragma unroll
            for (int j = 0; j < 4; ++j) { a[j] = (__bf16)pa[it][j]; b[j] = (__bf16)pb[it][j]; }
            *(v4bf*)&As[r][c4 * 4] = a;
            *(v4bf*)&Bs[r][c4 * 4] = b;
        }
    };

    globalLoad(0);
    for (int k0 = 0; k0 < Ec; k0 += BK) {
        ldsStore();
        __syncthreads();
        if (k0 + BK < Ec) globalLoad(k0 + BK);   // overlap with WMMA burst

        v16bf af[4];
#pragma unroll
        for (int mf = 0; mf < 4; ++mf)
            af[mf] = frag16(&As[wm * 64 + mf * 16 + lrow][0], o0, o1);
#pragma unroll
        for (int nf = 0; nf < 2; ++nf) {
            v16bf bf = frag16(&Bs[wn * 32 + nf * 16 + lrow][0], o0, o1);
#pragma unroll
            for (int mf = 0; mf < 4; ++mf)
                acc[nf][mf] = __builtin_amdgcn_wmma_f32_16x16x32_bf16(
                    false, af[mf], false, bf, (short)0, acc[nf][mf], false, false);
        }
        __syncthreads();
    }

    // Epilogue: per-column (part, head, d) mapping; 128-wide tile never crosses
    // a qkv-part boundary (1024 % 128 == 0).
#pragma unroll
    for (int nf = 0; nf < 2; ++nf) {
        const int f = f0 + wn * 32 + nf * 16 + lrow;
        const float bv = bias[f];
        const int part = f / Ec;
        const int e = f % Ec;
        const int h = e / Dc, d = e % Dc;
        __bf16* dst = (part == 0) ? Q : (part == 1 ? K : V);
        const float scl = (part == 0) ? 0.125f : 1.0f;  // fold 1/sqrt(64) into Q
#pragma unroll
        for (int mf = 0; mf < 4; ++mf) {
#pragma unroll
            for (int i = 0; i < 8; ++i) {
                int m = m0 + wm * 64 + mf * 16 + i + 8 * lhi;
                int bb = m / Sc, s = m % Sc;
                float val = (acc[nf][mf][i] + bv) * scl;
                dst[((size_t)(bb * Hc + h) * Sc + s) * Dc + d] = (__bf16)val;
            }
        }
    }
}

// ============================================================================
// Kernel 2: causal flash attention per (b,h). 4 waves x 32 queries = 128 q/block.
// K and V tiles staged once per block in LDS; Q register-resident.
// 16 WMMAs per 32-key iteration per wave.
// ============================================================================
__global__ __launch_bounds__(128) void attn_kernel(
    const __bf16* __restrict__ Q, const __bf16* __restrict__ Km,
    const __bf16* __restrict__ Vm, __bf16* __restrict__ O) {
    constexpr int LDT = 48, LDK = 72;                // 96B / 144B rows (16B aligned)
    __shared__ __align__(16) __bf16 Vs[Dc][LDT];     // transposed V tile [d][key]
    __shared__ __align__(16) __bf16 Ks[32][LDK];     // K tile [key][d]
    __shared__ __align__(16) __bf16 Ps[4][32][LDT];  // per-wave P bounce

    const int bh = blockIdx.y;
    const int b = bh / Hc, h = bh % Hc;
    const __bf16* Qp = Q  + (size_t)bh * Sc * Dc;
    const __bf16* Kp = Km + (size_t)bh * Sc * Dc;
    const __bf16* Vp = Vm + (size_t)bh * Sc * Dc;
    const int tid = threadIdx.x, lane = tid & 31, wv = tid >> 5;
    const int lrow = lane & 15, lhi = lane >> 4;
    const int o0 = lhi ? 8 : 0, o1 = lhi ? 24 : 16;
    const int q0 = blockIdx.x * 128 + wv * 32;

    // Resident Q: 2 sub-tiles of 16 queries x 64 dim, each = two K=32 A-frags.
    v16bf qa[2][2];
#pragma unroll
    for (int qq = 0; qq < 2; ++qq)
#pragma unroll
        for (int j = 0; j < 2; ++j)
            qa[qq][j] = frag16(Qp + (size_t)(q0 + qq * 16 + lrow) * Dc + j * 32, o0, o1);

    v8f acc[2][4];
    float mrow[2][8], lsum[2][8];
#pragma unroll
    for (int qq = 0; qq < 2; ++qq) {
#pragma unroll
        for (int n = 0; n < 4; ++n) acc[qq][n] = zero8();
#pragma unroll
        for (int i = 0; i < 8; ++i) { mrow[qq][i] = -3.0e38f; lsum[qq][i] = 0.0f; }
    }

    const int kend = blockIdx.x * 128 + 128;  // uniform causal bound per block
    for (int kt = 0; kt < kend; kt += 32) {
        __syncthreads();  // previous iteration done reading Ks/Vs
        // Stage K tile (32 keys x 64 d, straight copy) and V tile (transposed).
#pragma unroll
        for (int it = 0; it < 2; ++it) {
            int c = tid + it * 128;            // 256 x 16B chunks each
            int key = c >> 3, dp = (c & 7) * 8;
            *(v8bf*)&Ks[key][dp] = *(const v8bf*)(Kp + (size_t)(kt + key) * Dc + dp);
            v8bf vv = *(const v8bf*)(Vp + (size_t)(kt + key) * Dc + dp);
#pragma unroll
            for (int j = 0; j < 8; ++j) Vs[dp + j][key] = vv[j];
        }
        __syncthreads();

        // Scores: S[qq][g] = Q_qq * K_g^T (contraction over D=64).
        v8f sc[2][2];
#pragma unroll
        for (int g = 0; g < 2; ++g) {
#pragma unroll
            for (int qq = 0; qq < 2; ++qq) sc[qq][g] = zero8();
#pragma unroll
            for (int j = 0; j < 2; ++j) {
                v16bf kb = frag16(&Ks[g * 16 + lrow][j * 32], o0, o1);
#pragma unroll
                for (int qq = 0; qq < 2; ++qq)
                    sc[qq][g] = __builtin_amdgcn_wmma_f32_16x16x32_bf16(
                        false, qa[qq][j], false, kb, (short)0, sc[qq][g], false, false);
            }
        }
        // Causal mask + online softmax per query sub-tile.
#pragma unroll
        for (int qq = 0; qq < 2; ++qq) {
#pragma unroll
            for (int g = 0; g < 2; ++g) {
                int kcol = kt + g * 16 + lrow;
#pragma unroll
                for (int i = 0; i < 8; ++i)
                    if (kcol > q0 + qq * 16 + i + 8 * lhi) sc[qq][g][i] = -3.0e38f;
            }
            float tm[8];
#pragma unroll
            for (int i = 0; i < 8; ++i) tm[i] = fmaxf(sc[qq][0][i], sc[qq][1][i]);
#pragma unroll
            for (int msk = 1; msk < 16; msk <<= 1)
#pragma unroll
                for (int i = 0; i < 8; ++i) tm[i] = fmaxf(tm[i], __shfl_xor(tm[i], msk, 32));
            float alpha[8], rs[8];
#pragma unroll
            for (int i = 0; i < 8; ++i) {
                float mn = fmaxf(mrow[qq][i], tm[i]);
                alpha[i] = __expf(mrow[qq][i] - mn);
                mrow[qq][i] = mn;
                rs[i] = 0.0f;
            }
#pragma unroll
            for (int g = 0; g < 2; ++g)
#pragma unroll
                for (int i = 0; i < 8; ++i) {
                    float p = __expf(sc[qq][g][i] - mrow[qq][i]);
                    sc[qq][g][i] = p;
                    rs[i] += p;
                }
#pragma unroll
            for (int msk = 1; msk < 16; msk <<= 1)
#pragma unroll
                for (int i = 0; i < 8; ++i) rs[i] += __shfl_xor(rs[i], msk, 32);
#pragma unroll
            for (int i = 0; i < 8; ++i) lsum[qq][i] = lsum[qq][i] * alpha[i] + rs[i];
#pragma unroll
            for (int n = 0; n < 4; ++n)
#pragma unroll
                for (int i = 0; i < 8; ++i) acc[qq][n][i] *= alpha[i];
            // P: C-layout f32 -> bf16 A-layout via wave-private LDS bounce.
#pragma unroll
            for (int g = 0; g < 2; ++g)
#pragma unroll
                for (int i = 0; i < 8; ++i)
                    Ps[wv][qq * 16 + i + 8 * lhi][g * 16 + lrow] = (__bf16)sc[qq][g][i];
        }
        asm volatile("s_wait_dscnt 0" ::: "memory");  // wave-local LDS RAW
        v16bf pa[2];
#pragma unroll
        for (int qq = 0; qq < 2; ++qq)
            pa[qq] = frag16(&Ps[wv][qq * 16 + lrow][0], o0, o1);

        // O += P * V (contraction over 32 keys, 4 N-tiles across head dim).
#pragma unroll
        for (int n = 0; n < 4; ++n) {
            v16bf vb = frag16(&Vs[n * 16 + lrow][0], o0, o1);
#pragma unroll
            for (int qq = 0; qq < 2; ++qq)
                acc[qq][n] = __builtin_amdgcn_wmma_f32_16x16x32_bf16(
                    false, pa[qq], false, vb, (short)0, acc[qq][n], false, false);
        }
    }

    // Epilogue: normalize, write O in [B,S,H*D] (token-major) bf16.
#pragma unroll
    for (int qq = 0; qq < 2; ++qq)
#pragma unroll
        for (int n = 0; n < 4; ++n) {
            int d = n * 16 + lrow;
#pragma unroll
            for (int i = 0; i < 8; ++i) {
                int qi = q0 + qq * 16 + i + 8 * lhi;
                float o = acc[qq][n][i] / lsum[qq][i];
                O[(size_t)(b * Sc + qi) * Ec + h * Dc + d] = (__bf16)o;
            }
        }
}

// ============================================================================
// Kernel 3: out = O @ W_out^T + b_out (M=4096, N=1024, K=1024), f32 output.
// Same pipelined 128x128 scheme; A already bf16.
// ============================================================================
__global__ __launch_bounds__(256) void out_gemm_kernel(
    const __bf16* __restrict__ A, const float* __restrict__ W,
    const float* __restrict__ bias, float* __restrict__ out) {
    constexpr int BM = 128, BN = 128, BK = 32, LDT = 48;
    __shared__ __align__(16) __bf16 As[BM][LDT];
    __shared__ __align__(16) __bf16 Bs[BN][LDT];

    const int tid = threadIdx.x;
    const int lane = tid & 31, wv = tid >> 5;
    const int wm = wv >> 2, wn = wv & 3;
    const int m0 = blockIdx.y * BM;
    const int f0 = blockIdx.x * BN;
    const int lrow = lane & 15, lhi = lane >> 4;
    const int o0 = lhi ? 8 : 0, o1 = lhi ? 24 : 16;

    v8f acc[2][4];
#pragma unroll
    for (int nf = 0; nf < 2; ++nf)
#pragma unroll
        for (int mf = 0; mf < 4; ++mf) acc[nf][mf] = zero8();

    v8bf pa[2];            // A prefetch: 128x32 bf16 = 512 x 16B chunks / 256 thr
    v4f  pb[4];            // B prefetch: 128x32 f32  = 1024 float4 / 256 thr
    auto globalLoad = [&](int k0) {
#pragma unroll
        for (int it = 0; it < 2; ++it) {
            int idx = tid + it * 256;
            int r = idx >> 2, cp = (idx & 3) * 8;
            pa[it] = *(const v8bf*)(A + (size_t)(m0 + r) * Ec + k0 + cp);
        }
#pragma unroll
        for (int it = 0; it < 4; ++it) {
            int idx = tid + it * 256;
            int r = idx >> 3, c4 = idx & 7;
            pb[it] = *(const v4f*)(W + (size_t)(f0 + r) * Ec + k0 + c4 * 4);
        }
    };
    auto ldsStore = [&]() {
#pragma unroll
        for (int it = 0; it < 2; ++it) {
            int idx = tid + it * 256;
            int r = idx >> 2, cp = (idx & 3) * 8;
            *(v8bf*)&As[r][cp] = pa[it];
        }
#pragma unroll
        for (int it = 0; it < 4; ++it) {
            int idx = tid + it * 256;
            int r = idx >> 3, c4 = idx & 7;
            v4bf b;
#pragma unroll
            for (int j = 0; j < 4; ++j) b[j] = (__bf16)pb[it][j];
            *(v4bf*)&Bs[r][c4 * 4] = b;
        }
    };

    globalLoad(0);
    for (int k0 = 0; k0 < Ec; k0 += BK) {
        ldsStore();
        __syncthreads();
        if (k0 + BK < Ec) globalLoad(k0 + BK);

        v16bf af[4];
#pragma unroll
        for (int mf = 0; mf < 4; ++mf)
            af[mf] = frag16(&As[wm * 64 + mf * 16 + lrow][0], o0, o1);
#pragma unroll
        for (int nf = 0; nf < 2; ++nf) {
            v16bf bf = frag16(&Bs[wn * 32 + nf * 16 + lrow][0], o0, o1);
#pragma unroll
            for (int mf = 0; mf < 4; ++mf)
                acc[nf][mf] = __builtin_amdgcn_wmma_f32_16x16x32_bf16(
                    false, af[mf], false, bf, (short)0, acc[nf][mf], false, false);
        }
        __syncthreads();
    }

#pragma unroll
    for (int nf = 0; nf < 2; ++nf) {
        const int f = f0 + wn * 32 + nf * 16 + lrow;
        const float bv = bias[f];
#pragma unroll
        for (int mf = 0; mf < 4; ++mf) {
#pragma unroll
            for (int i = 0; i < 8; ++i) {
                int m = m0 + wm * 64 + mf * 16 + i + 8 * lhi;
                out[(size_t)m * Ec + f] = acc[nf][mf][i] + bv;
            }
        }
    }
}

// ============================================================================
extern "C" void kernel_launch(void* const* d_in, const int* in_sizes, int n_in,
                              void* d_out, int out_size, void* d_ws, size_t ws_size,
                              hipStream_t stream) {
    (void)in_sizes; (void)n_in; (void)out_size; (void)ws_size;
    const float* X     = (const float*)d_in[0];
    const float* W_in  = (const float*)d_in[1];
    const float* b_in  = (const float*)d_in[2];
    const float* W_out = (const float*)d_in[3];
    const float* b_out = (const float*)d_in[4];
    float* out = (float*)d_out;

    // Workspace: Q,K,V in [B,H,S,D] bf16 + O in [B,S,E] bf16 = 32 MB total.
    const size_t nElem = (size_t)Bc * Hc * Sc * Dc;  // 4 Mi elements each
    __bf16* Qw = (__bf16*)d_ws;
    __bf16* Kw = Qw + nElem;
    __bf16* Vw = Kw + nElem;
    __bf16* Ow = Vw + nElem;

    qkv_gemm_kernel<<<dim3(E3 / 128, (Bc * Sc) / 128), 256, 0, stream>>>(
        X, W_in, b_in, Qw, Kw, Vw);
    attn_kernel<<<dim3(Sc / 128, Bc * Hc), 128, 0, stream>>>(Qw, Kw, Vw, Ow);
    out_gemm_kernel<<<dim3(Ec / 128, (Bc * Sc) / 128), 256, 0, stream>>>(
        Ow, W_out, b_out, out);
}
